// SelfAttention_50525995270631
// MI455X (gfx1250) — matrix operand order repaired and checked by reading, not compiled, not run on hardware
//
#include <hip/hip_runtime.h>

// ---------------- types for WMMA ----------------
typedef __bf16 bf16_t;
typedef __attribute__((ext_vector_type(16))) __bf16 v16bf;
typedef __attribute__((ext_vector_type(8)))  __bf16 v8bf;
typedef __attribute__((ext_vector_type(8)))  float  v8f;

#define WMMA_BF16(a, b, c) \
  __builtin_amdgcn_wmma_f32_16x16x32_bf16(false, (a), false, (b), (short)0, (c), false, false)

// ---------------- problem constants ----------------
#define EMBED  1024
#define S_LEN  2048
#define BATCH  2
#define HEADS  16
#define HDIM   64
#define ROWS   (BATCH * S_LEN)   // 4096 token rows
#define N3E    (3 * EMBED)       // 3072
#define LORA_R 128

// f32 -> bf16 round-to-nearest-even
static __device__ __forceinline__ unsigned short f2bf(float f) {
  unsigned int u = __float_as_uint(f);
  unsigned int r = (u + 0x7FFFu + ((u >> 16) & 1u)) >> 16;
  return (unsigned short)r;
}

// Assemble a v16bf from two 16-byte segments (each lowers to a b128 load).
static __device__ __forceinline__ v16bf load16(const unsigned short* lo,
                                               const unsigned short* hi) {
  v8bf l = *(const v8bf*)lo;
  v8bf h = *(const v8bf*)hi;
  v16bf r;
#pragma unroll
  for (int i = 0; i < 8; i++) { r[i] = l[i]; r[8 + i] = h[i]; }
  return r;
}

// ---------------- prep kernels ----------------

__global__ void fold3_bf16_kernel(const float* __restrict__ in,
                                  unsigned short* __restrict__ out, int rows) {
  int i = blockIdx.x * blockDim.x + threadIdx.x;
  int total = rows * EMBED;
  if (i >= total) return;
  int n = i / EMBED, e = i - n * EMBED;
  const float* p = in + (size_t)n * N3E + e;
  out[i] = f2bf(p[0] + p[EMBED] + p[2 * EMBED]);
}

__global__ void cvt_bf16_kernel(const float* __restrict__ in,
                                unsigned short* __restrict__ out, int n) {
  int i = blockIdx.x * blockDim.x + threadIdx.x;
  if (i < n) out[i] = f2bf(in[i]);
}

// ---------------- bf16 NT GEMM: C = alpha * A(MxK) * Bt(NxK)^T [+ C] ----------------
// One wave computes a 32(M) x 64(N) tile: 2 A-fragments x 4 B-fragments = 8 WMMAs
// per K=32 step, with register double-buffering so next-step loads are issued
// before the current WMMA group (avoids s_wait_loadcnt 0 in front of every WMMA).
// Requirements: K % 32 == 0, M % 32 == 0, N % 64 == 0.
__global__ void __launch_bounds__(256)
gemm_bf16_nt_kernel(const unsigned short* __restrict__ A,
                    const unsigned short* __restrict__ Bt,
                    float* __restrict__ C,
                    int M, int N, int K, float alpha, int accum) {
  int wid = blockIdx.x * (blockDim.x >> 5) + (threadIdx.x >> 5);
  int tiles_m = M >> 5;
  int tm = wid % tiles_m;
  int tn = wid / tiles_m;
  if (tn >= (N >> 6)) return;           // uniform per-wave guard (EXEC stays all-1)

  int lane = threadIdx.x & 31;
  int r  = lane & 15;
  int hf = lane >> 4;
  int row0 = tm << 5, col0 = tn << 6;

  const unsigned short* arow0 = A + (size_t)(row0 + r) * K + 16 * hf;
  const unsigned short* arow1 = A + (size_t)(row0 + 16 + r) * K + 16 * hf;
  const unsigned short* brow  = Bt + (size_t)(col0 + r) * K + 16 * hf;
  const size_t bj = (size_t)16 * K;     // stride between B fragment columns

  v8f acc0[4] = {v8f{0}, v8f{0}, v8f{0}, v8f{0}};
  v8f acc1[4] = {v8f{0}, v8f{0}, v8f{0}, v8f{0}};

  // prologue: fragments for k0 = 0
  v16bf a0 = load16(arow0, arow0 + 8);
  v16bf a1 = load16(arow1, arow1 + 8);
  v16bf b0 = load16(brow,          brow + 8);
  v16bf b1 = load16(brow + bj,     brow + bj + 8);
  v16bf b2 = load16(brow + 2 * bj, brow + 2 * bj + 8);
  v16bf b3 = load16(brow + 3 * bj, brow + 3 * bj + 8);

  for (int k0 = 32; k0 < K; k0 += 32) {
    // issue next-step loads first, then consume current fragments
    v16bf a0n = load16(arow0 + k0, arow0 + k0 + 8);
    v16bf a1n = load16(arow1 + k0, arow1 + k0 + 8);
    v16bf b0n = load16(brow + k0,          brow + k0 + 8);
    v16bf b1n = load16(brow + bj + k0,     brow + bj + k0 + 8);
    v16bf b2n = load16(brow + 2 * bj + k0, brow + 2 * bj + k0 + 8);
    v16bf b3n = load16(brow + 3 * bj + k0, brow + 3 * bj + k0 + 8);

    acc0[0] = WMMA_BF16(a0, b0, acc0[0]);
    acc1[0] = WMMA_BF16(a1, b0, acc1[0]);
    acc0[1] = WMMA_BF16(a0, b1, acc0[1]);
    acc1[1] = WMMA_BF16(a1, b1, acc1[1]);
    acc0[2] = WMMA_BF16(a0, b2, acc0[2]);
    acc1[2] = WMMA_BF16(a1, b2, acc1[2]);
    acc0[3] = WMMA_BF16(a0, b3, acc0[3]);
    acc1[3] = WMMA_BF16(a1, b3, acc1[3]);

    a0 = a0n; a1 = a1n; b0 = b0n; b1 = b1n; b2 = b2n; b3 = b3n;
  }
  // final K step
  acc0[0] = WMMA_BF16(a0, b0, acc0[0]);
  acc1[0] = WMMA_BF16(a1, b0, acc1[0]);
  acc0[1] = WMMA_BF16(a0, b1, acc0[1]);
  acc1[1] = WMMA_BF16(a1, b1, acc1[1]);
  acc0[2] = WMMA_BF16(a0, b2, acc0[2]);
  acc1[2] = WMMA_BF16(a1, b2, acc1[2]);
  acc0[3] = WMMA_BF16(a0, b3, acc0[3]);
  acc1[3] = WMMA_BF16(a1, b3, acc1[3]);

#pragma unroll
  for (int v = 0; v < 8; v++) {
    float* c0 = C + (size_t)(row0 + v + 8 * hf) * N + col0 + r;
    float* c1 = C + (size_t)(row0 + 16 + v + 8 * hf) * N + col0 + r;
#pragma unroll
    for (int j = 0; j < 4; j++) {
      float o0 = alpha * acc0[j][v];
      float o1 = alpha * acc1[j][v];
      if (accum) { o0 += c0[16 * j]; o1 += c1[16 * j]; }
      c0[16 * j] = o0;
      c1[16 * j] = o1;
    }
  }
}

// ---------------- qkv split: f32 [4096][3072] -> Qbf/Kbf [B,H,S,D], Vt [B,H,D,S] ----------------
__global__ void qkv_split_kernel(const float* __restrict__ qkv,
                                 unsigned short* __restrict__ Qb,
                                 unsigned short* __restrict__ Kb,
                                 unsigned short* __restrict__ Vt) {
  int i = blockIdx.x * blockDim.x + threadIdx.x;
  if (i >= ROWS * N3E) return;
  int m = i / N3E, n = i - m * N3E;
  int b = m >> 11, s = m & 2047;
  float v = qkv[i];
  int part = n >> 10;       // 0=Q 1=K 2=V
  int e = n & 1023;
  int hd = e >> 6, d = e & 63;
  size_t hb = (size_t)(b * HEADS + hd);
  if (part == 0)      Qb[(hb * S_LEN + s) * HDIM + d] = f2bf(v * 0.125f); // fold 1/sqrt(64)
  else if (part == 1) Kb[(hb * S_LEN + s) * HDIM + d] = f2bf(v);
  else                Vt[(hb * HDIM + d) * S_LEN + s] = f2bf(v);
}

// ---------------- flash attention: one wave per 16-query tile ----------------
// LDS transpose buffer is private per wave -> no __syncthreads needed; a bare
// s_wait_dscnt 0 orders the intra-wave P store->load through LDS without
// draining the global-load pipeline.
__global__ void __launch_bounds__(256)
attn_kernel(const unsigned short* __restrict__ Qb,
            const unsigned short* __restrict__ Kb,
            const unsigned short* __restrict__ Vt,
            unsigned short* __restrict__ Ob) {
  __shared__ unsigned short sP[8][16 * 32];   // per-wave 16x32 bf16 transpose buffer

  int wv   = threadIdx.x >> 5;
  int wid  = blockIdx.x * 8 + wv;             // 0..4095, exact
  int lane = threadIdx.x & 31;
  int r  = lane & 15;
  int hf = lane >> 4;

  int qtile = wid & 127;                      // 128 query tiles per head
  int headi = wid >> 7;                       // b*16 + head
  int q0 = qtile << 4;

  const unsigned short* Qh = Qb + (size_t)headi * S_LEN * HDIM;
  const unsigned short* Kh = Kb + (size_t)headi * S_LEN * HDIM;
  const unsigned short* Vh = Vt + (size_t)headi * HDIM * S_LEN;

  // Q A-fragments (held in registers): dims 0-31 and 32-63
  const unsigned short* qrow = Qh + (size_t)(q0 + r) * HDIM;
  v16bf aq0 = load16(qrow + 0  + 16 * hf, qrow + 8 + 16 * hf);
  v16bf aq1 = load16(qrow + 32 + 16 * hf, qrow + 40 + 16 * hf);

  v8f o[4] = {v8f{0}, v8f{0}, v8f{0}, v8f{0}};
  float mrow[8], lrow[8];
#pragma unroll
  for (int v = 0; v < 8; v++) { mrow[v] = -3.402823e38f; lrow[v] = 0.0f; }

  // K fragment loader (Bt role = K rows [key][dim])
  auto loadK = [&](int kt, v16bf& f0a, v16bf& f0b, v16bf& f1a, v16bf& f1b) {
    const unsigned short* k0p = Kh + (size_t)(kt + r) * HDIM + 16 * hf;
    const unsigned short* k1p = Kh + (size_t)(kt + 16 + r) * HDIM + 16 * hf;
    f0a = load16(k0p, k0p + 8);            // dims 0..31 slice
    f0b = load16(k0p + 32, k0p + 40);      // dims 32..63 slice
    f1a = load16(k1p, k1p + 8);
    f1b = load16(k1p + 32, k1p + 40);
  };

  v16bf bk0a, bk0b, bk1a, bk1b;
  loadK(0, bk0a, bk0b, bk1a, bk1b);

  for (int kt = 0; kt < S_LEN; kt += 32) {
    // ---- scores for keys [kt, kt+32) ----
    v8f s0 = v8f{0}, s1 = v8f{0};
    s0 = WMMA_BF16(aq0, bk0a, s0);
    s0 = WMMA_BF16(aq1, bk0b, s0);
    s1 = WMMA_BF16(aq0, bk1a, s1);
    s1 = WMMA_BF16(aq1, bk1b, s1);

    // pipeline: next-iteration K fragments + this-iteration V fragments,
    // issued before the softmax VALU block hides their latency
    if (kt + 32 < S_LEN) {
      loadK(kt + 32, bk0a, bk0b, bk1a, bk1b);
      __builtin_prefetch(Kh + (size_t)(kt + 64 + r) * HDIM, 0, 1);
    }
    v16bf bv[4];
#pragma unroll
    for (int j = 0; j < 4; j++) {
      const unsigned short* vp = Vh + (size_t)(16 * j + r) * S_LEN + kt + 16 * hf;
      bv[j] = load16(vp, vp + 8);
    }

    // ---- online softmax (row reductions across the 16-lane half-groups) ----
    float corr[8];
#pragma unroll
    for (int v = 0; v < 8; v++) {
      float mx = fmaxf(s0[v], s1[v]);
#pragma unroll
      for (int msk = 1; msk < 16; msk <<= 1) mx = fmaxf(mx, __shfl_xor(mx, msk, 32));
      float mn = fmaxf(mrow[v], mx);
      float c  = __expf(mrow[v] - mn);
      float p0 = __expf(s0[v] - mn);
      float p1 = __expf(s1[v] - mn);
      float ls = p0 + p1;
#pragma unroll
      for (int msk = 1; msk < 16; msk <<= 1) ls += __shfl_xor(ls, msk, 32);
      lrow[v] = lrow[v] * c + ls;
      mrow[v] = mn;
      corr[v] = c;
      s0[v] = p0;
      s1[v] = p1;
    }
#pragma unroll
    for (int j = 0; j < 4; j++)
#pragma unroll
      for (int v = 0; v < 8; v++) o[j][v] *= corr[v];

    // ---- transpose P (C-layout -> A-layout) through wave-private LDS ----
    unsigned short* pw = sP[wv];
#pragma unroll
    for (int v = 0; v < 8; v++) {
      pw[(v + 8 * hf) * 32 + r]      = f2bf(s0[v]);
      pw[(v + 8 * hf) * 32 + 16 + r] = f2bf(s1[v]);
    }
    asm volatile("s_wait_dscnt 0x0" ::: "memory");   // intra-wave LDS WAR/RAW order
    v16bf ap = load16(pw + r * 32 + 16 * hf, pw + r * 32 + 8 + 16 * hf);

    // ---- P(16x32) @ V(32x64) via Vt rows [dim][key] ----
#pragma unroll
    for (int j = 0; j < 4; j++) o[j] = WMMA_BF16(ap, bv[j], o[j]);
  }

  // ---- finalize: divide by l, store bf16 to attn_out [ROWS][EMBED] ----
  int b  = headi >> 4;
  int hd = headi & 15;
#pragma unroll
  for (int v = 0; v < 8; v++) {
    float inv = 1.0f / lrow[v];
    size_t row = (size_t)(b * S_LEN + q0 + v + 8 * hf);
    unsigned short* orow = Ob + row * EMBED + hd * HDIM + r;
#pragma unroll
    for (int j = 0; j < 4; j++) orow[16 * j] = f2bf(o[j][v] * inv);
  }
}

// ---------------- host side ----------------
static inline int gemm_blocks(int M, int N) {
  int waves = (M >> 5) * (N >> 6);    // 32x64 tile per wave
  return (waves + 7) / 8;             // 8 waves (256 threads) per block
}

extern "C" void kernel_launch(void* const* d_in, const int* in_sizes, int n_in,
                              void* d_out, int out_size, void* d_ws, size_t ws_size,
                              hipStream_t stream) {
  const float* x     = (const float*)d_in[0];   // [2,2048,1024]
  const float* W_in  = (const float*)d_in[1];   // [3072,3072]
  const float* A_in  = (const float*)d_in[2];   // [128,3072]
  const float* B_in  = (const float*)d_in[3];   // [3072,128]
  const float* W_out = (const float*)d_in[4];   // [1024,1024]
  float* out = (float*)d_out;                   // [2,2048,1024]

  char* ws = (char*)d_ws;
  size_t off = 0;
  auto alloc = [&](size_t bytes) -> char* {
    char* p = ws + off;
    off += (bytes + 255) & ~(size_t)255;
    return p;
  };

  float* qkv_f  = (float*)alloc((size_t)ROWS * N3E * 4);        // 48 MB
  float* lora_f = (float*)alloc((size_t)ROWS * LORA_R * 4);     // 2 MB
  unsigned short* x_bf    = (unsigned short*)alloc((size_t)ROWS * EMBED * 2);
  unsigned short* Weff    = (unsigned short*)alloc((size_t)N3E * EMBED * 2);
  unsigned short* Aeff    = (unsigned short*)alloc((size_t)LORA_R * EMBED * 2);
  unsigned short* Bbf     = (unsigned short*)alloc((size_t)N3E * LORA_R * 2);
  unsigned short* Wobf    = (unsigned short*)alloc((size_t)EMBED * EMBED * 2);
  unsigned short* lora_bf = (unsigned short*)alloc((size_t)ROWS * LORA_R * 2);
  unsigned short* Qb      = (unsigned short*)alloc((size_t)ROWS * EMBED * 2);
  unsigned short* Kb      = (unsigned short*)alloc((size_t)ROWS * EMBED * 2);
  unsigned short* Vt      = (unsigned short*)alloc((size_t)ROWS * EMBED * 2);
  unsigned short* Ob      = (unsigned short*)alloc((size_t)ROWS * EMBED * 2);

  const int T = 256;
  // 1) precision prep / weight folding
  cvt_bf16_kernel<<<(ROWS * EMBED + T - 1) / T, T, 0, stream>>>(x, x_bf, ROWS * EMBED);
  fold3_bf16_kernel<<<(N3E * EMBED + T - 1) / T, T, 0, stream>>>(W_in, Weff, N3E);
  fold3_bf16_kernel<<<(LORA_R * EMBED + T - 1) / T, T, 0, stream>>>(A_in, Aeff, LORA_R);
  cvt_bf16_kernel<<<(N3E * LORA_R + T - 1) / T, T, 0, stream>>>(B_in, Bbf, N3E * LORA_R);
  cvt_bf16_kernel<<<(EMBED * EMBED + T - 1) / T, T, 0, stream>>>(W_out, Wobf, EMBED * EMBED);

  // 2) qkv = x @ Weff^T
  gemm_bf16_nt_kernel<<<gemm_blocks(ROWS, N3E), T, 0, stream>>>(
      x_bf, Weff, qkv_f, ROWS, N3E, EMBED, 1.0f, 0);
  // 3) lora = x @ Aeff^T ; qkv += (lora @ B_in^T) / 128
  gemm_bf16_nt_kernel<<<gemm_blocks(ROWS, LORA_R), T, 0, stream>>>(
      x_bf, Aeff, lora_f, ROWS, LORA_R, EMBED, 1.0f, 0);
  cvt_bf16_kernel<<<(ROWS * LORA_R + T - 1) / T, T, 0, stream>>>(lora_f, lora_bf, ROWS * LORA_R);
  gemm_bf16_nt_kernel<<<gemm_blocks(ROWS, N3E), T, 0, stream>>>(
      lora_bf, Bbf, qkv_f, ROWS, N3E, LORA_R, 1.0f / 128.0f, 1);

  // 4) split into Q (pre-scaled), K, V^T in attention layout
  qkv_split_kernel<<<(ROWS * N3E + T - 1) / T, T, 0, stream>>>(qkv_f, Qb, Kb, Vt);

  // 5) flash attention: 4096 waves = 512 blocks x 8 waves
  attn_kernel<<<512, T, 0, stream>>>(Qb, Kb, Vt, Ob);

  // 6) out = attn_out @ W_out^T (f32 result straight to d_out)
  gemm_bf16_nt_kernel<<<gemm_blocks(ROWS, EMBED), T, 0, stream>>>(
      Ob, Wobf, out, ROWS, EMBED, EMBED, 1.0f, 0);
  (void)in_sizes; (void)n_in; (void)out_size; (void)ws_size;
}
